// RFRM_alpha_70918499992036
// MI455X (gfx1250) — matrix operand order repaired and checked by reading, not compiled, not run on hardware
//
#include <hip/hip_runtime.h>
#include <hip/hip_bf16.h>
#include <math.h>

// Problem constants (from reference setup_inputs)
#define Nn   2
#define Cc   256
#define Hh   96
#define Ww   96
#define HW   (Hh * Ww)          // 9216
#define Mtot (Nn * HW)          // 18432 pixels
#define Ktot (Cc * 9)           // 2304 contraction dim
#define Oo   256

typedef __attribute__((ext_vector_type(16))) __bf16 v16bf;
typedef __attribute__((ext_vector_type(8)))  __bf16 v8bf;
typedef __attribute__((ext_vector_type(8)))  float  v8f;

static __device__ __forceinline__ unsigned short f2bf(float f) {
    // round-to-nearest-even fp32 -> bf16
    unsigned int u = __float_as_uint(f);
    unsigned int lsb = (u >> 16) & 1u;
    u += 0x7fffu + lsb;
    return (unsigned short)(u >> 16);
}

// ---------------------------------------------------------------------------
// Kernel 1: offset-generating convs (tm: 4ch, tr: 2ch) -> dy/dx per tap.
// One thread per pixel.
// ---------------------------------------------------------------------------
__global__ void offsets_kernel(const float* __restrict__ x,
                               const float* __restrict__ w_tm,
                               const float* __restrict__ b_tm,
                               const float* __restrict__ w_tr,
                               const float* __restrict__ b_tr,
                               float* __restrict__ dyArr,
                               float* __restrict__ dxArr) {
    int p = blockIdx.x * blockDim.x + threadIdx.x;
    if (p >= Mtot) return;
    int n = p / HW, hw = p % HW, h = hw / Ww, w = hw % Ww;

    float a0 = b_tm[0], a1 = b_tm[1], a2 = b_tm[2], a3 = b_tm[3];
    float a4 = b_tr[0], a5 = b_tr[1];

    for (int c = 0; c < Cc; ++c) {
        const float* xc = x + (size_t)(n * Cc + c) * HW;
        #pragma unroll
        for (int t = 0; t < 9; ++t) {
            int ky = t / 3, kx = t % 3;
            int y  = h + ky - 1, xx = w + kx - 1;
            if (y < 0 || y >= Hh || xx < 0 || xx >= Ww) continue;
            float v  = xc[y * Ww + xx];
            int   wi = c * 9 + t;
            a0 = fmaf(v, w_tm[0 * Ktot + wi], a0);
            a1 = fmaf(v, w_tm[1 * Ktot + wi], a1);
            a2 = fmaf(v, w_tm[2 * Ktot + wi], a2);
            a3 = fmaf(v, w_tm[3 * Ktot + wi], a3);
            a4 = fmaf(v, w_tr[0 * Ktot + wi], a4);
            a5 = fmaf(v, w_tr[1 * Ktot + wi], a5);
        }
    }
    // T = [[a0,a1],[a2,a3]]; off[i][k] = sum_j T[i][j]*REG[j][k] - REG[i][k] + tr[i]
    #pragma unroll
    for (int k = 0; k < 9; ++k) {
        float ry = (float)(k / 3) - 1.0f;   // REG[0][k]
        float rx = (float)(k % 3) - 1.0f;   // REG[1][k]
        dyArr[k * Mtot + p] = a0 * ry + a1 * rx - ry + a4;
        dxArr[k * Mtot + p] = a2 * ry + a3 * rx - rx + a5;
    }
}

// ---------------------------------------------------------------------------
// Kernel 2: weight prep.  B[o][k*256+c] = bf16( w_deform[o][c][k] )
// ---------------------------------------------------------------------------
__global__ void wprep_kernel(const float* __restrict__ w_deform,
                             unsigned short* __restrict__ Bm) {
    int e = blockIdx.x * blockDim.x + threadIdx.x;
    if (e >= Oo * Ktot) return;
    int o = e / Ktot, r = e % Ktot, c = r / 9, k = r % 9;
    Bm[(size_t)o * Ktot + k * Cc + c] = f2bf(w_deform[e]);
}

// ---------------------------------------------------------------------------
// Kernel 3: deformable im2col.  Block = (pixel, tap); threads over c.
// A[p][k*256+c] = bf16( bilinear_sample(x[n,c], py, px) )
// ---------------------------------------------------------------------------
__global__ void im2col_kernel(const float* __restrict__ x,
                              const float* __restrict__ dyArr,
                              const float* __restrict__ dxArr,
                              unsigned short* __restrict__ Am) {
    int b = blockIdx.x;         // in [0, Mtot*9)
    int p = b / 9, k = b % 9;
    int c = threadIdx.x;
    int n = p / HW, hw = p % HW, h = hw / Ww, w = hw % Ww;

    float dy = dyArr[k * Mtot + p];
    float dx = dxArr[k * Mtot + p];
    float py = dy + (float)(h - 1 + (k / 3));
    float px = dx + (float)(w - 1 + (k % 3));
    float y0f = floorf(py), x0f = floorf(px);
    float wy = py - y0f, wx = px - x0f;
    int y0 = (int)y0f, x0 = (int)x0f;

    const float* xc = x + (size_t)(n * Cc + c) * HW;

    auto g = [&](int yi, int xi) -> float {
        bool ok = (yi >= 0) && (yi < Hh) && (xi >= 0) && (xi < Ww);
        int yc  = min(max(yi, 0), Hh - 1);
        int xcl = min(max(xi, 0), Ww - 1);
        float v = xc[yc * Ww + xcl];
        return ok ? v : 0.0f;
    };

    float s = g(y0,     x0    ) * (1.0f - wy) * (1.0f - wx)
            + g(y0,     x0 + 1) * (1.0f - wy) * wx
            + g(y0 + 1, x0    ) * wy * (1.0f - wx)
            + g(y0 + 1, x0 + 1) * wy * wx;

    Am[(size_t)p * Ktot + k * Cc + c] = f2bf(s);
}

// ---------------------------------------------------------------------------
// Kernel 4: register-blocked WMMA GEMM  D[p,o] = sum_K A[p,K]*B[o,K]
// with fused BN + residual + ReLU epilogue.
//
// 256 threads = 8 waves arranged 4(M) x 2(N); block tile 128(M) x 128(N).
// Each wave computes a 32x64 strip = 2(M) x 4(N) subtiles of 16x16:
//   8 v_wmma_f32_16x16x32_bf16 per K-step on 12 b128 loads (fragment reuse).
// ---------------------------------------------------------------------------
__global__ void __launch_bounds__(256)
gemm_kernel(const unsigned short* __restrict__ Am,
            const unsigned short* __restrict__ Bm,
            const float* __restrict__ x,
            const float* __restrict__ gamma,
            const float* __restrict__ beta,
            const float* __restrict__ mean,
            const float* __restrict__ var,
            float* __restrict__ out) {
    const int wave = threadIdx.x >> 5;
    const int lane = threadIdx.x & 31;
    const int wm   = wave & 3;              // 0..3  (M direction)
    const int wn   = wave >> 2;             // 0..1  (N direction)
    const int bn   = blockIdx.x & 1;        // 2 N-blocks of 128
    const int bm   = blockIdx.x >> 1;       // 144 M-blocks of 128
    const int half = lane >> 4;             // 0: lanes 0-15, 1: lanes 16-31
    const int r    = lane & 15;

    const int m0 = bm * 128 + wm * 32;      // wave's first output row
    const int n0 = bn * 128 + wn * 64;      // wave's first output column

    // A-fragment rows (16-bit A 16x32 ISA layout: lane = row r, two 8-elem
    // chunks at kk + half*8 and kk + 16 + half*8).
    const unsigned short* arow[2];
    #pragma unroll
    for (int i = 0; i < 2; ++i)
        arow[i] = Am + (size_t)(m0 + i * 16 + r) * Ktot;

    // B-fragment rows (lane = column r, two 8-elem chunks at kk + half*8 and kk + 16 + half*8).
    const unsigned short* brow[4];
    #pragma unroll
    for (int j = 0; j < 4; ++j)
        brow[j] = Bm + (size_t)(n0 + j * 16 + r) * Ktot;

    v8f acc[2][4] = {};

    for (int kk = 0; kk < Ktot; kk += 32) {
        v16bf a[2], b[4];

        #pragma unroll
        for (int i = 0; i < 2; ++i) {
            v8bf lo = *(const v8bf*)(arow[i] + kk + half * 8);
            v8bf hi = *(const v8bf*)(arow[i] + kk + 16 + half * 8);
            #pragma unroll
            for (int e = 0; e < 8; ++e) { a[i][e] = lo[e]; a[i][e + 8] = hi[e]; }
        }
        #pragma unroll
        for (int j = 0; j < 4; ++j) {
            v8bf lo = *(const v8bf*)(brow[j] + kk + half * 8);
            v8bf hi = *(const v8bf*)(brow[j] + kk + 16 + half * 8);
            #pragma unroll
            for (int e = 0; e < 8; ++e) { b[j][e] = lo[e]; b[j][e + 8] = hi[e]; }
        }

        // Unguarded speculative prefetch of the next K block of A (WGP scope;
        // past-the-end prefetch is silently dropped).
        __builtin_prefetch(arow[0] + kk + 32, 0, 3);
        __builtin_prefetch(arow[1] + kk + 32, 0, 3);

        #pragma unroll
        for (int i = 0; i < 2; ++i)
            #pragma unroll
            for (int j = 0; j < 4; ++j)
                acc[i][j] = __builtin_amdgcn_wmma_f32_16x16x32_bf16(
                                false, a[i], false, b[j],
                                (short)0, acc[i][j], false, false);
    }

    // Epilogue: BN + residual + ReLU, NCHW scatter.
    #pragma unroll
    for (int j = 0; j < 4; ++j) {
        const int o = n0 + j * 16 + r;                 // lane's output channel
        const float sc = gamma[o] * rsqrtf(var[o] + 1e-5f);
        const float sh = beta[o] - mean[o] * sc;
        #pragma unroll
        for (int i = 0; i < 2; ++i) {
            #pragma unroll
            for (int jj = 0; jj < 8; ++jj) {
                int p  = m0 + i * 16 + half * 8 + jj;  // C/D layout: M = half*8+jj
                int n  = p / HW, hw = p % HW;
                size_t idx = (size_t)(n * Cc + o) * HW + hw;
                float v = acc[i][j][jj] * sc + sh + x[idx];
                out[idx] = v > 0.0f ? v : 0.0f;
            }
        }
    }
}

// ---------------------------------------------------------------------------
// Host-side launcher
// ---------------------------------------------------------------------------
extern "C" void kernel_launch(void* const* d_in, const int* in_sizes, int n_in,
                              void* d_out, int out_size, void* d_ws, size_t ws_size,
                              hipStream_t stream) {
    const float* x        = (const float*)d_in[0];
    // d_in[1] = mask_attention (unused by reference)
    const float* w_tm     = (const float*)d_in[2];
    const float* b_tm     = (const float*)d_in[3];
    const float* w_tr     = (const float*)d_in[4];
    const float* b_tr     = (const float*)d_in[5];
    const float* w_deform = (const float*)d_in[6];
    const float* bn_gamma = (const float*)d_in[7];
    const float* bn_beta  = (const float*)d_in[8];
    const float* bn_mean  = (const float*)d_in[9];
    const float* bn_var   = (const float*)d_in[10];
    float* out = (float*)d_out;

    // Workspace layout (256B-aligned slabs):
    //   dyArr : 9*Mtot f32      (663552 B)
    //   dxArr : 9*Mtot f32      (663552 B)
    //   Bmat  : Oo*Ktot bf16    (1179648 B)
    //   Amat  : Mtot*Ktot bf16  (84934656 B)
    char* ws = (char*)d_ws;
    float*          dyArr = (float*)(ws);
    float*          dxArr = (float*)(ws + 663552);
    unsigned short* Bmat  = (unsigned short*)(ws + 1327104);
    unsigned short* Amat  = (unsigned short*)(ws + 2506752);

    // 1) offset field
    offsets_kernel<<<(Mtot + 255) / 256, 256, 0, stream>>>(
        x, w_tm, b_tm, w_tr, b_tr, dyArr, dxArr);

    // 2) weight transpose + bf16
    wprep_kernel<<<(Oo * Ktot + 255) / 256, 256, 0, stream>>>(w_deform, Bmat);

    // 3) deformable im2col (bilinear gather)
    im2col_kernel<<<Mtot * 9, 256, 0, stream>>>(x, dyArr, dxArr, Amat);

    // 4) register-blocked WMMA GEMM + fused BN/residual/ReLU
    //    grid = (Mtot/128) * (Oo/128) = 144 * 2 = 288 blocks of 8 waves
    gemm_kernel<<<(Mtot / 128) * (Oo / 128), 256, 0, stream>>>(
        Amat, Bmat, x, bn_gamma, bn_beta, bn_mean, bn_var, out);
}
